// Model_51384988729796
// MI455X (gfx1250) — matrix operand order repaired and checked by reading, not compiled, not run on hardware
//
#include <hip/hip_runtime.h>
#include <hip/hip_bf16.h>

typedef _Float16 h16;
typedef __attribute__((ext_vector_type(16))) _Float16 v16h;
typedef __attribute__((ext_vector_type(8)))  _Float16 v8h;
typedef __attribute__((ext_vector_type(8)))  float    v8f;
typedef unsigned int u32;

#define NEG_SLOPE (-0.8f)

__device__ __forceinline__ float lrelu(float x) { return x > 0.f ? x : NEG_SLOPE * x; }

// ---------------------------------------------------------------------------
// Utility kernels
// ---------------------------------------------------------------------------
__global__ void k_zero32(u32* p, long n) {
    long i = (long)blockIdx.x * blockDim.x + threadIdx.x;
    if (i < n) p[i] = 0u;
}

// count in-degree
__global__ void k_count(const int* __restrict__ dst, u32* __restrict__ deg, int E) {
    int i = blockIdx.x * blockDim.x + threadIdx.x;
    if (i < E) atomicAdd(&deg[dst[i]], 1u);
}

// per-block inclusive scan of deg -> rp[i+1]; block totals to bsums
__global__ void k_scan_block(const u32* __restrict__ deg, u32* __restrict__ rp,
                             u32* __restrict__ bsums, int n) {
    __shared__ u32 sh[256];
    int t = threadIdx.x;
    int i = blockIdx.x * 256 + t;
    u32 v = (i < n) ? deg[i] : 0u;
    sh[t] = v;
    __syncthreads();
    for (int off = 1; off < 256; off <<= 1) {
        u32 add = (t >= off) ? sh[t - off] : 0u;
        __syncthreads();
        sh[t] += add;
        __syncthreads();
    }
    if (i < n) rp[i + 1] = sh[t];
    if (t == 255) bsums[blockIdx.x] = sh[255];
}

// exclusive scan of block sums (few hundred elements; single thread is fine)
__global__ void k_scan_top(u32* bsums, int nb) {
    u32 run = 0;
    for (int i = 0; i < nb; i++) { u32 v = bsums[i]; bsums[i] = run; run += v; }
}

__global__ void k_scan_add(u32* __restrict__ rp, const u32* __restrict__ bsums, int n) {
    int i = blockIdx.x * blockDim.x + threadIdx.x;
    if (i < n) rp[i + 1] += bsums[i >> 8];
    if (i == 0) rp[0] = 0u;
}

// scatter edge ids into CSR (cursor = zeroed deg buffer reuse)
__global__ void k_fill(const int* __restrict__ dst, u32* __restrict__ cursor,
                       const u32* __restrict__ rp, u32* __restrict__ csr, int E) {
    int i = blockIdx.x * blockDim.x + threadIdx.x;
    if (i >= E) return;
    int d = dst[i];
    u32 pos = atomicAdd(&cursor[d], 1u);
    csr[rp[d] + pos] = (u32)i;
}

// convert+pad f32 [N,ncols] -> f16 [Nalloc,dcols] (zeros elsewhere)
__global__ void k_prep_pad(const float* __restrict__ src, int ncols,
                           h16* __restrict__ dst, int dcols, int N, long total) {
    long i = (long)blockIdx.x * blockDim.x + threadIdx.x;
    if (i >= total) return;
    int n = (int)(i / dcols), j = (int)(i % dcols);
    float v = (n < N && j < ncols) ? src[(long)n * ncols + j] : 0.f;
    dst[i] = (h16)v;
}

// weight prep: W f32 [I,O] row-major -> Wt f16 [opad, kpad] (transposed, padded).
// perm!=0: storage row o holds logical output column (o&7)*32 + (o>>3), so the
// GEMM result comes out in [F=32, K=8] interleaved order (one contiguous v8h of
// K-values per feature) for the aggregation gather.
__global__ void k_prep_w(const float* __restrict__ W, int I, int O,
                         h16* __restrict__ Wt, int kpad, int opad, int perm) {
    int i = blockIdx.x * blockDim.x + threadIdx.x;
    int total = kpad * opad;
    if (i >= total) return;
    int o = i / kpad, k = i % kpad;
    int ol = perm ? ((o & 7) * 32 + (o >> 3)) : o;
    float v = (k < I && ol < O) ? W[(long)k * O + ol] : 0.f;
    Wt[i] = (h16)v;
}

// final copy: f16 [N,16] cols 0..4 -> f32 d_out [N,5]
__global__ void k_out(const h16* __restrict__ p, float* __restrict__ out, int N) {
    int i = blockIdx.x * blockDim.x + threadIdx.x;
    if (i >= N * 5) return;
    int n = i / 5, c = i % 5;
    out[i] = (float)p[(long)n * 16 + c];
}

// ---------------------------------------------------------------------------
// WMMA GEMM:  Out[N16, opad](f16) = act( A[N16, kpad](f16) @ Wt^T + bias )
// Wt is [opad, kpad] f16 (pre-transposed, zero-padded).  One wave per 16x16
// output tile; 4 waves/block cover 4 column tiles.
// ---------------------------------------------------------------------------
__global__ __launch_bounds__(128) void k_gemm(const h16* __restrict__ A, int kpad,
                                              const h16* __restrict__ Wt,
                                              const float* __restrict__ bias, int O,
                                              h16* __restrict__ Out, int opad,
                                              int do_lrelu, int perm) {
    int lane = threadIdx.x;                       // 0..31
    int ct = blockIdx.y * blockDim.y + threadIdx.y;
    if (ct * 16 >= opad) return;                  // wave-uniform exit
    int rt = blockIdx.x;
    int hi = lane >> 4;                           // lane half (0/1)
    int l15 = lane & 15;
    int row = rt * 16 + l15;                      // A row owned by this lane
    int colg = ct * 16 + l15;                     // B column owned by this lane
    const h16* Arow = A + (size_t)row * kpad;
    const h16* Brow = Wt + (size_t)colg * kpad;

    v8f c;
    #pragma unroll
    for (int i = 0; i < 8; i++) c[i] = 0.f;

    for (int kb = 0; kb < kpad; kb += 32) {
        // A 16x32 f16 layout: halves 0..7 -> K = kb + 8*hi .. +7
        //                     halves 8..15 -> K = kb + 16 + 8*hi .. +7
        v8h a0 = *(const v8h*)(Arow + kb + 8 * hi);
        v8h a1 = *(const v8h*)(Arow + kb + 16 + 8 * hi);
        v16h a = __builtin_shufflevector(a0, a1, 0, 1, 2, 3, 4, 5, 6, 7,
                                         8, 9, 10, 11, 12, 13, 14, 15);
        // B 32x16 f16 layout: lane = column, halves 0..7 -> K = kb + 8*hi .. +7, halves 8..15 -> K = kb + 16 + 8*hi .. +7
        v8h b0 = *(const v8h*)(Brow + kb + 8 * hi);
        v8h b1 = *(const v8h*)(Brow + kb + 16 + 8 * hi);
        v16h b = __builtin_shufflevector(b0, b1, 0, 1, 2, 3, 4, 5, 6, 7,
                                         8, 9, 10, 11, 12, 13, 14, 15);
        c = __builtin_amdgcn_wmma_f32_16x16x32_f16(false, a, false, b,
                                                   (short)0, c, false, false);
    }

    int bcol = perm ? ((colg & 7) * 32 + (colg >> 3)) : colg;   // logical column
    float bv = (bcol < O) ? bias[bcol] : 0.f;
    #pragma unroll
    for (int r = 0; r < 8; r++) {
        int grow = rt * 16 + r + 8 * hi;          // D: VGPR r -> M = r + 8*hi
        float v = c[r] + bv;
        if (do_lrelu) v = lrelu(v);
        Out[(size_t)grow * opad + colg] = (h16)v;
    }
}

// ---------------------------------------------------------------------------
// Edge aggregation: one wave per destination node.
//   rst[n,f]  = sum_{e: dst=n} sum_k gauss(pseudo[e];mu_k)   * hk[src[e], f, k]
//   rst_e[n]  = (sum_{e} sum_k gauss(pseudo[e];mu_e_k) * he[src[e],k]) / max(deg,1)
// hk is stored [N, F=32, K=8] so each lane's 8 K-values are ONE b128 load and
// the 32 lanes read a fully contiguous 512B row (L2-resident).  Lanes 0-7
// compute node-kernel gaussians, lanes 8-15 energy-kernel gaussians; broadcast
// with wave32 __shfl.  Next source row is prefetched (global_prefetch_b8).
// ---------------------------------------------------------------------------
__global__ __launch_bounds__(256) void k_agg(
    const u32* __restrict__ rp, const u32* __restrict__ csr,
    const int* __restrict__ esrc, const float* __restrict__ pseudo,
    const h16* __restrict__ hk, const h16* __restrict__ heK,
    const float* __restrict__ mu, const float* __restrict__ invs,
    const float* __restrict__ mue, const float* __restrict__ invse,
    const float* __restrict__ bias,
    h16* __restrict__ hout, h16* __restrict__ eout, h16* __restrict__ ycat,
    int layer, int N) {
    int wid = (blockIdx.x * blockDim.x + threadIdx.x) >> 5;   // node id
    int lane = threadIdx.x & 31;
    if (wid >= N) return;                                     // wave-uniform

    int kk = lane & 7;
    const float* M = (lane & 8) ? mue : mu;
    const float* S = (lane & 8) ? invse : invs;
    float m[4], s2[4];
    #pragma unroll
    for (int d = 0; d < 4; d++) {
        m[d] = M[kk * 4 + d];
        float s = S[kk * 4 + d];
        s2[d] = s * s;
    }

    u32 beg = rp[wid], end = rp[wid + 1];
    float acc = 0.f, esum = 0.f;
    int f = lane;

    for (u32 i = beg; i < end; i++) {
        int e = (int)csr[i];
        int s = esrc[e];
        if (i + 1 < end) {                         // prefetch next source row
            int sn = esrc[(int)csr[i + 1]];
            __builtin_prefetch(hk + (size_t)sn * 256 + (size_t)f * 8, 0, 1);
        }
        float4 uv = ((const float4*)pseudo)[e];
        float d0 = uv.x - m[0], d1 = uv.y - m[1], d2 = uv.z - m[2], d3 = uv.w - m[3];
        float g = __expf(-0.5f * (d0 * d0 * s2[0] + d1 * d1 * s2[1] +
                                  d2 * d2 * s2[2] + d3 * d3 * s2[3]));
        // one contiguous 16B load: this lane's 8 K-values for feature f
        v8h hv = *(const v8h*)(hk + (size_t)s * 256 + (size_t)f * 8);
        #pragma unroll
        for (int k = 0; k < 8; k++) {
            float wk = __shfl(g, k, 32);                      // node gaussian k
            acc += wk * (float)hv[k];
        }
        float wek = __shfl(g, 8 + kk, 32);                    // energy gaussian
        if (lane < 8) esum += wek * (float)heK[(size_t)s * 16 + lane];
    }

    hout[(size_t)wid * 32 + f] = (h16)(acc + bias[f]);

    for (int off = 16; off; off >>= 1) esum += __shfl_down(esum, off, 32);
    if (lane == 0) {
        float deg = (float)(end - beg);
        float re = esum / fmaxf(deg, 1.f);
        eout[(size_t)wid * 32] = (h16)re;                     // col0; rest pre-zeroed
        ycat[(size_t)wid * 32 + layer] = (h16)re;
    }
}

// ---------------------------------------------------------------------------
// Host orchestration
// ---------------------------------------------------------------------------
extern "C" void kernel_launch(void* const* d_in, const int* in_sizes, int n_in,
                              void* d_out, int out_size, void* d_ws, size_t ws_size,
                              hipStream_t stream) {
    const float* x      = (const float*)d_in[0];
    const float* x_en   = (const float*)d_in[1];
    const float* pseudo = (const float*)d_in[2];
    const int*   esrc   = (const int*)d_in[3];
    const int*   edst   = (const int*)d_in[4];
    const int N = in_sizes[0] / 3;
    const int E = in_sizes[3];
    const int N16 = (N + 15) & ~15;

    // bump allocator over workspace
    char* ws = (char*)d_ws;
    size_t off = 0;
    auto alloc = [&](size_t bytes) -> void* {
        off = (off + 255) & ~(size_t)255;
        void* p = ws + off;
        off += bytes;
        return p;
    };
    h16* hA   = (h16*)alloc((size_t)N16 * 32 * 2);
    h16* hB   = (h16*)alloc((size_t)N16 * 32 * 2);
    h16* heA  = (h16*)alloc((size_t)N16 * 32 * 2);
    h16* heB  = (h16*)alloc((size_t)N16 * 32 * 2);
    h16* t0   = (h16*)alloc((size_t)N16 * 32 * 2);
    h16* t1   = (h16*)alloc((size_t)N16 * 64 * 2);
    h16* t2   = (h16*)alloc((size_t)N16 * 64 * 2);
    h16* hk   = (h16*)alloc((size_t)N16 * 256 * 2);
    h16* heK  = (h16*)alloc((size_t)N16 * 16 * 2);
    h16* ycat = (h16*)alloc((size_t)N16 * 32 * 2);
    h16* Wt   = (h16*)alloc((size_t)256 * 64 * 2);
    u32* deg  = (u32*)alloc((size_t)N * 4);
    u32* rp   = (u32*)alloc((size_t)(N + 1) * 4);
    u32* bsum = (u32*)alloc((size_t)4096 * 4);
    u32* csr  = (u32*)alloc((size_t)E * 4);
    (void)ws_size; (void)n_in; (void)out_size;

    const int T = 256;
    auto blks = [](long n, int t) { return (unsigned)((n + t - 1) / t); };

    // ---- CSR build (by destination) ----
    k_zero32<<<blks(N, T), T, 0, stream>>>(deg, N);
    k_count<<<blks(E, T), T, 0, stream>>>(edst, deg, E);
    int nb = (N + 255) / 256;
    k_scan_block<<<nb, 256, 0, stream>>>(deg, rp, bsum, N);
    k_scan_top<<<1, 1, 0, stream>>>(bsum, nb);
    k_scan_add<<<blks(N, T), T, 0, stream>>>(rp, bsum, N);
    k_zero32<<<blks(N, T), T, 0, stream>>>(deg, N);           // cursor
    k_fill<<<blks(E, T), T, 0, stream>>>(edst, deg, rp, csr, E);

    // ---- input prep ----
    long tot32 = (long)N16 * 32;
    k_prep_pad<<<blks(tot32, T), T, 0, stream>>>(x, 3, hA, 32, N, tot32);
    k_prep_pad<<<blks(tot32, T), T, 0, stream>>>(x_en, 1, heA, 32, N, tot32);
    k_zero32<<<blks(tot32 / 2, T), T, 0, stream>>>((u32*)heB, tot32 / 2);
    k_zero32<<<blks(tot32 / 2, T), T, 0, stream>>>((u32*)ycat, tot32 / 2);

    // GEMM launcher: prep weights then run WMMA GEMM
    auto gemm = [&](const h16* Ain, int kpad, const float* W, const float* b,
                    int I, int O, h16* Outp, int opad, int dol, int perm) {
        int wtot = opad * kpad;
        k_prep_w<<<blks(wtot, T), T, 0, stream>>>(W, I, O, Wt, kpad, opad, perm);
        dim3 blk(32, 4);
        dim3 grd(N16 / 16, (opad + 63) / 64);
        k_gemm<<<grd, blk, 0, stream>>>(Ain, kpad, Wt, b, O, Outp, opad, dol, perm);
    };

    // ---- 4 GMMConv layers ----
    for (int l = 0; l < 4; l++) {
        int base = 5 + l * 21;
        const float* fcW[4], *fcb[4], *feW[4], *feb[4];
        for (int j = 0; j < 4; j++) {
            fcW[j] = (const float*)d_in[base + j];
            fcb[j] = (const float*)d_in[base + 4 + j];
            feW[j] = (const float*)d_in[base + 8 + j];
            feb[j] = (const float*)d_in[base + 12 + j];
        }
        const float* mu    = (const float*)d_in[base + 16];
        const float* invs  = (const float*)d_in[base + 17];
        const float* mue   = (const float*)d_in[base + 18];
        const float* invse = (const float*)d_in[base + 19];
        const float* bias  = (const float*)d_in[base + 20];

        h16* hin  = (l & 1) ? hB : hA;
        h16* hout = (l & 1) ? hA : hB;
        h16* ein  = (l & 1) ? heB : heA;
        h16* eout = (l & 1) ? heA : heB;
        int in_f = (l == 0) ? 3 : 32;

        // node MLP: in_f -> 20 -> 40 -> 60 -> 256   (all LeakyReLU)
        gemm(hin, 32, fcW[0], fcb[0], in_f, 20, t0, 32, 1, 0);
        gemm(t0, 32, fcW[1], fcb[1], 20, 40, t1, 64, 1, 0);
        gemm(t1, 64, fcW[2], fcb[2], 40, 60, t2, 64, 1, 0);
        gemm(t2, 64, fcW[3], fcb[3], 60, 256, hk, 256, 1, 1);   // [F,K] permuted
        // energy MLP: 1 -> 20 -> 40 -> 60 -> 8      (all LeakyReLU)
        gemm(ein, 32, feW[0], feb[0], 1, 20, t0, 32, 1, 0);
        gemm(t0, 32, feW[1], feb[1], 20, 40, t1, 64, 1, 0);
        gemm(t1, 64, feW[2], feb[2], 40, 60, t2, 64, 1, 0);
        gemm(t2, 64, feW[3], feb[3], 60, 8, heK, 16, 1, 0);
        // edge gather + kernel-weighted aggregation
        k_agg<<<blks((long)N * 32, 256), 256, 0, stream>>>(
            rp, csr, esrc, pseudo, hk, heK, mu, invs, mue, invse, bias,
            hout, eout, ycat, l, N);
    }

    // ---- projection MLP: 4 -> 20 -> 40 -> 30 -> 10 -> 5 -> 5 ----
    const float* pW[6], *pb[6];
    for (int j = 0; j < 6; j++) {
        pW[j] = (const float*)d_in[89 + j];
        pb[j] = (const float*)d_in[95 + j];
    }
    gemm(ycat, 32, pW[0], pb[0], 4, 20, t1, 32, 1, 0);
    gemm(t1, 32, pW[1], pb[1], 20, 40, t2, 64, 1, 0);
    gemm(t2, 64, pW[2], pb[2], 40, 30, t1, 32, 1, 0);
    gemm(t1, 32, pW[3], pb[3], 30, 10, t2, 32, 1, 0);
    gemm(t2, 32, pW[4], pb[4], 10, 5, t1, 32, 1, 0);
    gemm(t1, 32, pW[5], pb[5], 5, 5, t2, 16, 0, 0);   // final: no activation

    k_out<<<blks((long)N * 5, T), T, 0, stream>>>(t2, (float*)d_out, N);
}
